// convcap_60954175865356
// MI455X (gfx1250) — matrix operand order repaired and checked by reading, not compiled, not run on hardware
//
#include <hip/hip_runtime.h>
#include <hip/hip_bf16.h>
#include <math.h>

// ---------------------------------------------------------------------------
// Types for CDNA5 WMMA (wave32): v_wmma_f32_16x16x32_bf16
// ---------------------------------------------------------------------------
typedef __bf16 bf16;
typedef __attribute__((ext_vector_type(16))) __bf16 v16bf;
typedef __attribute__((ext_vector_type(8)))  __bf16 bf16x8;
typedef __attribute__((ext_vector_type(8)))  float   v8f;

#define SQ05F 0.70710678118654752440f

__device__ __forceinline__ v16bf zero16() {
  v16bf z;
#pragma unroll
  for (int i = 0; i < 16; ++i) z[i] = (__bf16)0.0f;
  return z;
}

__device__ __forceinline__ v8f zero8f() {
  v8f z;
#pragma unroll
  for (int i = 0; i < 8; ++i) z[i] = 0.0f;
  return z;
}

// Two contiguous 16B loads -> one 8-VGPR bf16 fragment.
__device__ __forceinline__ v16bf load2x8(const bf16* p0, const bf16* p1) {
  v16bf r;
  reinterpret_cast<bf16x8*>(&r)[0] = *reinterpret_cast<const bf16x8*>(p0);
  reinterpret_cast<bf16x8*>(&r)[1] = *reinterpret_cast<const bf16x8*>(p1);
  return r;
}

__device__ __forceinline__ v8f wmma_bf16(v16bf a, v16bf b, v8f c) {
  return __builtin_amdgcn_wmma_f32_16x16x32_bf16(false, a, false, b, (short)0, c,
                                                 false, false);
}

// ---------------------------------------------------------------------------
// Weight / data conversion kernels (fp32 -> transposed, padded bf16)
// ---------------------------------------------------------------------------

// W (fin,fout) fp32 -> Wt (Npad, fin) bf16, Wt[n][k] = W[k][n], pad rows = 0.
__global__ void k_cvt_lin(const float* __restrict__ W, bf16* __restrict__ Wt,
                          int fin, int fout, int Npad) {
  int idx = blockIdx.x * blockDim.x + threadIdx.x;
  int total = Npad * fin;
  if (idx >= total) return;
  int n = idx / fin, c = idx - n * fin;
  Wt[idx] = (n < fout) ? (bf16)W[(size_t)c * fout + n] : (bf16)0.0f;
}

// conv w (Cout,Cin,5) fp32 -> Wt (5,Cout,Cin) bf16
__global__ void k_cvt_conv(const float* __restrict__ W, bf16* __restrict__ Wt,
                           int Cout, int Cin) {
  int idx = blockIdx.x * blockDim.x + threadIdx.x;
  int total = 5 * Cout * Cin;
  if (idx >= total) return;
  int k = idx / (Cout * Cin);
  int r = idx - k * (Cout * Cin);
  int co = r / Cin, ci = r - co * Cin;
  Wt[idx] = (bf16)W[((size_t)co * Cin + ci) * 5 + k];
}

__global__ void k_cvt_f32bf(const float* __restrict__ X, bf16* __restrict__ Y,
                            int total) {
  int idx = blockIdx.x * blockDim.x + threadIdx.x;
  if (idx >= total) return;
  Y[idx] = (bf16)X[idx];
}

// E[r][c] = emb0[wordclass[r]][c]   (8192 x 512) bf16
__global__ void k_gather(const float* __restrict__ emb0,
                         const int* __restrict__ wc, bf16* __restrict__ E) {
  int idx = blockIdx.x * blockDim.x + threadIdx.x;
  if (idx >= 8192 * 512) return;
  int r = idx >> 9, c = idx & 511;
  E[idx] = (bf16)emb0[(size_t)wc[r] * 512 + c];
}

// X0[m][c] : c<512 -> wordemb[m][c] ; else y[m/32][c-512]   (8192 x 1024) bf16
__global__ void k_x0(const float* __restrict__ wordemb,
                     const float* __restrict__ y, bf16* __restrict__ X0) {
  int idx = blockIdx.x * blockDim.x + threadIdx.x;
  if (idx >= 8192 * 1024) return;
  int m = idx >> 10, c = idx & 1023;
  float v = (c < 512) ? wordemb[(size_t)m * 512 + c]
                      : y[(size_t)(m >> 5) * 512 + (c - 512)];
  X0[idx] = (bf16)v;
}

// yflatT[b][s][f] = s<49 ? img[b][f][s] : 0   (256,64,512) bf16  (Bt for scores)
__global__ void k_yflatT(const float* __restrict__ img, bf16* __restrict__ yT) {
  int idx = blockIdx.x * blockDim.x + threadIdx.x;
  if (idx >= 256 * 64 * 512) return;
  int b = idx / (64 * 512);
  int r = idx - b * (64 * 512);
  int s = r >> 9, f = r & 511;
  yT[idx] = (s < 49) ? (bf16)img[((size_t)b * 512 + f) * 49 + s] : (bf16)0.0f;
}

// yflatP[b][f][s] = s<49 ? img[b][f][s] : 0   (256,512,64) bf16  (Bt for context)
__global__ void k_yflatP(const float* __restrict__ img, bf16* __restrict__ yP) {
  int idx = blockIdx.x * blockDim.x + threadIdx.x;
  if (idx >= 256 * 512 * 64) return;
  int b = idx / (512 * 64);
  int r = idx - b * (512 * 64);
  int f = r >> 6, s = r & 63;
  yP[idx] = (s < 49) ? (bf16)img[((size_t)b * 512 + f) * 49 + s] : (bf16)0.0f;
}

// GLU over conv output (8192 x 1024) -> (8192 x 512) f32 + bf16
__global__ void k_glu(const float* __restrict__ Y, float* __restrict__ X,
                      bf16* __restrict__ Xbf) {
  int idx = blockIdx.x * blockDim.x + threadIdx.x;
  if (idx >= 8192 * 512) return;
  int m = idx >> 9, c = idx & 511;
  float a = Y[(size_t)m * 1024 + c];
  float g = Y[(size_t)m * 1024 + 512 + c];
  float v = a * (1.0f / (1.0f + __expf(-g)));
  X[idx] = v;
  Xbf[idx] = (bf16)v;
}

// o = (a + b) * SQ05  (+ bf16 mirror)
__global__ void k_addscale(const float* __restrict__ a,
                           const float* __restrict__ b, float* __restrict__ o,
                           bf16* __restrict__ obf, int total) {
  int idx = blockIdx.x * blockDim.x + threadIdx.x;
  if (idx >= total) return;
  float v = (a[idx] + b[idx]) * SQ05F;
  o[idx] = v;
  obf[idx] = (bf16)v;
}

// Wave-per-row softmax over 49 of 64 score columns.
// Writes attn (ld 49, fp32, final output region) and attnP (ld 64, bf16, padded).
__global__ void __launch_bounds__(256) k_softmax49(
    const float* __restrict__ scores, float* __restrict__ attn,
    bf16* __restrict__ attnP, int rows) {
  int w = blockIdx.x * 8 + (threadIdx.x >> 5);
  if (w >= rows) return;
  int lane = threadIdx.x & 31;
  const float* s = scores + (size_t)w * 64;
  float v0 = (lane < 49) ? s[lane] : -1e30f;
  float v1 = (lane + 32 < 49) ? s[lane + 32] : -1e30f;
  float m = fmaxf(v0, v1);
#pragma unroll
  for (int off = 16; off; off >>= 1) m = fmaxf(m, __shfl_xor(m, off, 32));
  float e0 = (lane < 49) ? __expf(v0 - m) : 0.0f;
  float e1 = (lane + 32 < 49) ? __expf(v1 - m) : 0.0f;
  float sum = e0 + e1;
#pragma unroll
  for (int off = 16; off; off >>= 1) sum += __shfl_xor(sum, off, 32);
  float inv = 1.0f / sum;
  float a0 = e0 * inv, a1 = e1 * inv;
  if (lane < 49) attn[(size_t)w * 49 + lane] = a0;
  if (lane + 32 < 49) attn[(size_t)w * 49 + lane + 32] = a1;
  attnP[(size_t)w * 64 + lane] = (bf16)a0;
  attnP[(size_t)w * 64 + lane + 32] = (bf16)a1;
}

// ---------------------------------------------------------------------------
// Big GEMM: each wave computes a 64x32 C macro-tile (4x2 WMMA 16x16 tiles).
// A (M x K) bf16 row-major, Bt (Npad x K) bf16 row-major (B transposed).
// C = act( (A@B + bias + res) * scale ); optional fp32 and/or bf16 outputs.
// ---------------------------------------------------------------------------
__global__ void __launch_bounds__(256) wgemm_big(
    const bf16* __restrict__ A, int lda, const bf16* __restrict__ Bt, int ldb,
    float* __restrict__ C, bf16* __restrict__ Cbf, int ldc,
    const float* __restrict__ bias, const float* __restrict__ res, float scale,
    int relu, int M, int N, int K, int tilesN, int totalWaves) {
  int w = blockIdx.x * 8 + (threadIdx.x >> 5);
  if (w >= totalWaves) return;          // wave-uniform: EXEC stays all-ones
  int lane = threadIdx.x & 31;
  int half = lane >> 4;
  int l16 = lane & 15;
  int tm = w / tilesN, tn = w - tm * tilesN;
  int baseM = tm * 64, baseN = tn * 32;

  v8f acc[4][2];
#pragma unroll
  for (int i = 0; i < 4; ++i)
#pragma unroll
    for (int j = 0; j < 2; ++j) acc[i][j] = zero8f();

  int ak0 = half ? 8 : 0;    // A: lanes 16-31 start at K=8 within the tile
  int bk0 = half ? 8 : 0;    // B: lanes 16-31 start at K=8 within the tile

  for (int kk = 0; kk < K; kk += 32) {
    v16bf af[4], bfr[2];
#pragma unroll
    for (int i = 0; i < 4; ++i) {
      const bf16* p = A + (size_t)(baseM + i * 16 + l16) * lda + kk + ak0;
      af[i] = load2x8(p, p + 16);
    }
#pragma unroll
    for (int j = 0; j < 2; ++j) {
      const bf16* p = Bt + (size_t)(baseN + j * 16 + l16) * ldb + kk + bk0;
      bfr[j] = load2x8(p, p + 16);
    }
    if (kk + 32 < K) {  // hint next K-slab into cache (global_prefetch_b8)
      __builtin_prefetch(A + (size_t)(baseM + l16) * lda + kk + 32, 0, 1);
      __builtin_prefetch(Bt + (size_t)(baseN + l16) * ldb + kk + 32, 0, 1);
    }
#pragma unroll
    for (int i = 0; i < 4; ++i)
#pragma unroll
      for (int j = 0; j < 2; ++j) acc[i][j] = wmma_bf16(af[i], bfr[j], acc[i][j]);
  }

#pragma unroll
  for (int i = 0; i < 4; ++i) {
#pragma unroll
    for (int j = 0; j < 2; ++j) {
      int n = baseN + j * 16 + l16;
      bool nok = n < N;
      float bv = (bias && nok) ? bias[n] : 0.0f;
#pragma unroll
      for (int r = 0; r < 8; ++r) {
        int m = baseM + i * 16 + half * 8 + r;
        if (nok && m < M) {
          float v = acc[i][j][r] + bv;
          if (res) v += res[(size_t)m * ldc + n];
          v *= scale;
          if (relu) v = v > 0.0f ? v : 0.0f;
          if (C) C[(size_t)m * ldc + n] = v;
          if (Cbf) Cbf[(size_t)m * ldc + n] = (bf16)v;
        }
      }
    }
  }
}

// ---------------------------------------------------------------------------
// Causal conv1d (K=5) as 5 shifted GEMMs, fused in one kernel.
// A (M x Cin) bf16 rows = (b,t); Wt (5, Cout, Cin) bf16.
// Y[m][co] = sum_k sum_ci A[m + k - 4][ci] * Wt[k][co][ci]   (masked t+k-4>=0)
// ---------------------------------------------------------------------------
__global__ void __launch_bounds__(256) wconv_big(
    const bf16* __restrict__ A, int Cin, const bf16* __restrict__ Wt,
    float* __restrict__ C, const float* __restrict__ bias, int M, int Cout,
    int tilesN, int totalWaves) {
  int w = blockIdx.x * 8 + (threadIdx.x >> 5);
  if (w >= totalWaves) return;
  int lane = threadIdx.x & 31;
  int half = lane >> 4;
  int l16 = lane & 15;
  int tm = w / tilesN, tn = w - tm * tilesN;
  int baseM = tm * 64, baseN = tn * 32;
  int ak0 = half ? 8 : 0;
  int bk0 = half ? 8 : 0;

  v8f acc[4][2];
#pragma unroll
  for (int i = 0; i < 4; ++i)
#pragma unroll
    for (int j = 0; j < 2; ++j) acc[i][j] = zero8f();

  for (int k = 0; k < 5; ++k) {
    int s = k - 4;  // time shift (<=0)
    const bf16* Bk = Wt + (size_t)k * Cout * Cin;
    for (int kk = 0; kk < Cin; kk += 32) {
      v16bf af[4], bfr[2];
#pragma unroll
      for (int i = 0; i < 4; ++i) {
        int arow = baseM + i * 16 + l16;
        bool valid = ((arow & 31) + s) >= 0;  // causal mask within T=32
        const bf16* p = A + (size_t)(arow + s) * Cin + kk + ak0;
        af[i] = valid ? load2x8(p, p + 16) : zero16();
      }
#pragma unroll
      for (int j = 0; j < 2; ++j) {
        const bf16* p = Bk + (size_t)(baseN + j * 16 + l16) * Cin + kk + bk0;
        bfr[j] = load2x8(p, p + 16);
      }
#pragma unroll
      for (int i = 0; i < 4; ++i)
#pragma unroll
        for (int j = 0; j < 2; ++j)
          acc[i][j] = wmma_bf16(af[i], bfr[j], acc[i][j]);
    }
  }

#pragma unroll
  for (int i = 0; i < 4; ++i) {
#pragma unroll
    for (int j = 0; j < 2; ++j) {
      int n = baseN + j * 16 + l16;
      float bv = bias[n];
#pragma unroll
      for (int r = 0; r < 8; ++r) {
        int m = baseM + i * 16 + half * 8 + r;
        C[(size_t)m * Cout + n] = acc[i][j][r] + bv;
      }
    }
  }
}

// ---------------------------------------------------------------------------
// Batched small GEMM: one 16x16 WMMA tile per wave (attention score/context).
// ---------------------------------------------------------------------------
__global__ void __launch_bounds__(256) wgemm_attn(
    const bf16* __restrict__ A, int lda, int sA, const bf16* __restrict__ Bt,
    int ldb, int sB, float* __restrict__ C, bf16* __restrict__ Cbf, int ldc,
    int sC, float alpha, int mtiles, int ntiles, int N, int K, int totalWaves) {
  int w = blockIdx.x * 8 + (threadIdx.x >> 5);
  if (w >= totalWaves) return;
  int lane = threadIdx.x & 31;
  int half = lane >> 4;
  int l16 = lane & 15;
  int tpb = mtiles * ntiles;
  int b = w / tpb;
  int rem = w - b * tpb;
  int mt = rem / ntiles, nt = rem - mt * ntiles;
  const bf16* Ab = A + (size_t)b * sA;
  const bf16* Bb = Bt + (size_t)b * sB;
  int ak0 = half ? 8 : 0;
  int bk0 = half ? 8 : 0;

  v8f acc = zero8f();
  for (int kk = 0; kk < K; kk += 32) {
    const bf16* pa = Ab + (size_t)(mt * 16 + l16) * lda + kk + ak0;
    const bf16* pb = Bb + (size_t)(nt * 16 + l16) * ldb + kk + bk0;
    v16bf af = load2x8(pa, pa + 16);
    v16bf bfr = load2x8(pb, pb + 16);
    acc = wmma_bf16(af, bfr, acc);
  }

  int n = nt * 16 + l16;
  if (n < N) {
#pragma unroll
    for (int r = 0; r < 8; ++r) {
      int m = mt * 16 + half * 8 + r;
      float v = acc[r] * alpha;
      size_t o = (size_t)b * sC + (size_t)m * ldc + n;
      if (C) C[o] = v;
      if (Cbf) Cbf[o] = (bf16)v;
    }
  }
}

// ---------------------------------------------------------------------------
// Host orchestration
// ---------------------------------------------------------------------------
extern "C" void kernel_launch(void* const* d_in, const int* in_sizes, int n_in,
                              void* d_out, int out_size, void* d_ws,
                              size_t ws_size, hipStream_t stream) {
  (void)in_sizes; (void)n_in; (void)out_size; (void)ws_size;
  const int B = 256, T = 32, M = 8192, NF = 512, V = 10000, NI = 4096;
  const int VPAD = 10016;

  const float* imgsfeat = (const float*)d_in[0];
  const float* imgsfc7  = (const float*)d_in[1];
  const int*   wordcls  = (const int*)d_in[2];
  const float* emb0_w   = (const float*)d_in[3];
  const float* emb1_w   = (const float*)d_in[4];
  const float* emb1_b   = (const float*)d_in[5];
  const float* iproj_w  = (const float*)d_in[6];
  const float* iproj_b  = (const float*)d_in[7];
  const float* rproj_w  = (const float*)d_in[8];
  const float* rproj_b  = (const float*)d_in[9];
  const float* conv0_w  = (const float*)d_in[10];
  const float* conv0_b  = (const float*)d_in[11];
  const float* convs_w  = (const float*)d_in[12];
  const float* convs_b  = (const float*)d_in[13];
  const float* ain_w    = (const float*)d_in[14];
  const float* ain_b    = (const float*)d_in[15];
  const float* aout_w   = (const float*)d_in[16];
  const float* aout_b   = (const float*)d_in[17];
  const float* cls0_w   = (const float*)d_in[18];
  const float* cls0_b   = (const float*)d_in[19];
  const float* cls1_w   = (const float*)d_in[20];
  const float* cls1_b   = (const float*)d_in[21];

  float* logits  = (float*)d_out;                       // (8192, 10000)
  float* attnOut = (float*)d_out + (size_t)M * V;       // (8192, 49)

  // ---- workspace bump allocator -------------------------------------------
  char* ws = (char*)d_ws;
  size_t off = 0;
  auto alloc = [&](size_t bytes) -> void* {
    void* r = ws + off;
    off = (off + bytes + 255) & ~(size_t)255;
    return r;
  };
  auto abf = [&](size_t elems) { return (bf16*)alloc(elems * sizeof(bf16)); };
  auto af32 = [&](size_t elems) { return (float*)alloc(elems * sizeof(float)); };

  bf16* emb1T   = abf((size_t)NF * NF);
  bf16* iprojT  = abf((size_t)NF * NI);
  bf16* rprojT  = abf((size_t)NF * 2 * NF);
  bf16* conv0T  = abf((size_t)5 * 1024 * 1024);
  bf16* convsT  = abf((size_t)2 * 5 * 1024 * 512);
  bf16* ainT    = abf((size_t)3 * NF * NF);
  bf16* aoutT   = abf((size_t)3 * NF * NF);
  bf16* cls0T   = abf((size_t)256 * NF);
  bf16* cls1T   = abf((size_t)VPAD * 256);
  bf16* E       = abf((size_t)M * NF);
  bf16* fc7bf   = abf((size_t)B * NI);
  float* wemb   = af32((size_t)M * NF);
  bf16* wembBf  = abf((size_t)M * NF);
  float* yproj  = af32((size_t)B * NF);
  bf16* X0bf    = abf((size_t)M * 2 * NF);
  float* res0   = af32((size_t)M * NF);
  bf16* yflatT  = abf((size_t)B * 64 * NF);
  bf16* yflatP  = abf((size_t)B * NF * 64);
  float* convY  = af32((size_t)M * 2 * NF);
  float* Xg     = af32((size_t)M * NF);
  bf16* Xgbf    = abf((size_t)M * NF);
  bf16* xabf    = abf((size_t)M * NF);
  float* scores = af32((size_t)M * 64);
  bf16* attnP   = abf((size_t)M * 64);
  bf16* ctxbf   = abf((size_t)M * NF);
  float* xtout  = af32((size_t)M * NF);
  float* Xf[2]  = {af32((size_t)M * NF), af32((size_t)M * NF)};
  bf16* Xb[2]   = {abf((size_t)M * NF), abf((size_t)M * NF)};
  bf16* hbf     = abf((size_t)M * 256);

  auto grid1 = [](int total) { return dim3((unsigned)((total + 255) / 256)); };

  auto gemm = [&](const bf16* A, int lda, const bf16* Bt, int ldb, float* C,
                  bf16* Cbf, int ldc, const float* bias, const float* res,
                  float scale, int relu, int Mr, int N, int Npad, int K) {
    int tilesN = Npad / 32;
    int waves = (Mr / 64) * tilesN;
    wgemm_big<<<dim3((unsigned)((waves + 7) / 8)), dim3(256), 0, stream>>>(
        A, lda, Bt, ldb, C, Cbf, ldc, bias, res, scale, relu, Mr, N, K, tilesN,
        waves);
  };

  // ---- 1. weight conversions ----------------------------------------------
  k_cvt_lin<<<grid1(NF * NF), 256, 0, stream>>>(emb1_w, emb1T, NF, NF, NF);
  k_cvt_lin<<<grid1(NF * NI), 256, 0, stream>>>(iproj_w, iprojT, NI, NF, NF);
  k_cvt_lin<<<grid1(NF * 1024), 256, 0, stream>>>(rproj_w, rprojT, 1024, NF, NF);
  k_cvt_conv<<<grid1(5 * 1024 * 1024), 256, 0, stream>>>(conv0_w, conv0T, 1024, 1024);
  for (int i = 0; i < 2; ++i)
    k_cvt_conv<<<grid1(5 * 1024 * 512), 256, 0, stream>>>(
        convs_w + (size_t)i * 1024 * 512 * 5, convsT + (size_t)i * 5 * 1024 * 512,
        1024, 512);
  for (int i = 0; i < 3; ++i) {
    k_cvt_lin<<<grid1(NF * NF), 256, 0, stream>>>(
        ain_w + (size_t)i * NF * NF, ainT + (size_t)i * NF * NF, NF, NF, NF);
    k_cvt_lin<<<grid1(NF * NF), 256, 0, stream>>>(
        aout_w + (size_t)i * NF * NF, aoutT + (size_t)i * NF * NF, NF, NF, NF);
  }
  k_cvt_lin<<<grid1(256 * NF), 256, 0, stream>>>(cls0_w, cls0T, NF, 256, 256);
  k_cvt_lin<<<grid1(VPAD * 256), 256, 0, stream>>>(cls1_w, cls1T, 256, V, VPAD);
  k_cvt_f32bf<<<grid1(B * NI), 256, 0, stream>>>(imgsfc7, fc7bf, B * NI);

  // ---- 2. embedding + projections -----------------------------------------
  k_gather<<<grid1(M * NF), 256, 0, stream>>>(emb0_w, wordcls, E);
  // wordemb = E @ emb1 + b
  gemm(E, NF, emb1T, NF, wemb, wembBf, NF, emb1_b, nullptr, 1.0f, 0, M, NF, NF, NF);
  // y = relu(fc7 @ imgproj + b)
  gemm(fc7bf, NI, iprojT, NI, yproj, nullptr, NF, iproj_b, nullptr, 1.0f, 1, B,
       NF, NF, NI);
  k_x0<<<grid1(M * 2 * NF), 256, 0, stream>>>(wemb, yproj, X0bf);
  // res0 = X0 @ resproj + b
  gemm(X0bf, 2 * NF, rprojT, 2 * NF, res0, nullptr, NF, rproj_b, nullptr, 1.0f,
       0, M, NF, NF, 2 * NF);
  k_yflatT<<<grid1(B * 64 * NF), 256, 0, stream>>>(imgsfeat, yflatT);
  k_yflatP<<<grid1(B * NF * 64), 256, 0, stream>>>(imgsfeat, yflatP);

  // ---- 3. conv + attention layers -----------------------------------------
  const bf16* convA = X0bf;
  const float* resF = res0;
  int cur = 0;
  for (int i = 0; i < 3; ++i) {
    int Cin = (i == 0) ? 1024 : 512;
    const bf16* Wt = (i == 0) ? conv0T : convsT + (size_t)(i - 1) * 5 * 1024 * 512;
    const float* cb = (i == 0) ? conv0_b : convs_b + (size_t)(i - 1) * 1024;
    {
      int tilesN = 1024 / 32;
      int waves = (M / 64) * tilesN;
      wconv_big<<<dim3((unsigned)((waves + 7) / 8)), dim3(256), 0, stream>>>(
          convA, Cin, Wt, convY, cb, M, 1024, tilesN, waves);
    }
    k_glu<<<grid1(M * NF), 256, 0, stream>>>(convY, Xg, Xgbf);

    // xa = (Xg @ in_w + in_b + wordemb) * SQ05     (bf16 only)
    gemm(Xgbf, NF, ainT + (size_t)i * NF * NF, NF, nullptr, xabf, NF,
         ain_b + (size_t)i * NF, wemb, SQ05F, 0, M, NF, NF, NF);

    // scores[b] = xa[b] @ yflatT[b]^T  : M=32,N=49(pad64),K=512
    {
      int waves = B * 2 * 4;
      wgemm_attn<<<dim3((unsigned)((waves + 7) / 8)), dim3(256), 0, stream>>>(
          xabf, NF, T * NF, yflatT, NF, 64 * NF, scores, nullptr, 64, T * 64,
          1.0f, 2, 4, 64, NF, waves);
    }
    k_softmax49<<<dim3((unsigned)(M / 8)), dim3(256), 0, stream>>>(scores, attnOut,
                                                                   attnP, M);
    // ctx = 7 * attn[b] @ yflatP[b]^T  : M=32,N=512,K=64
    {
      int waves = B * 2 * 32;
      wgemm_attn<<<dim3((unsigned)((waves + 7) / 8)), dim3(256), 0, stream>>>(
          attnP, 64, T * 64, yflatP, 64, NF * 64, nullptr, ctxbf, NF, T * NF,
          7.0f, 2, 32, NF, 64, waves);
    }
    // xt_out = (ctx @ out_w + out_b + Xg) * SQ05
    gemm(ctxbf, NF, aoutT + (size_t)i * NF * NF, NF, xtout, nullptr, NF,
         aout_b + (size_t)i * NF, Xg, SQ05F, 0, M, NF, NF, NF);
    // x = (xt_out + residual) * SQ05
    k_addscale<<<grid1(M * NF), 256, 0, stream>>>(xtout, resF, Xf[cur], Xb[cur],
                                                  M * NF);
    convA = Xb[cur];
    resF = Xf[cur];
    cur ^= 1;
  }

  // ---- 4. classifier -------------------------------------------------------
  const bf16* Xfin = convA;  // last layer's bf16 activations
  gemm(Xfin, NF, cls0T, NF, nullptr, hbf, 256, cls0_b, nullptr, 1.0f, 0, M, 256,
       256, NF);
  gemm(hbf, 256, cls1T, 256, logits, nullptr, V, cls1_b, nullptr, 1.0f, 0, M, V,
       VPAD, 256);
}
